// GraphEmbedding_11836929868230
// MI455X (gfx1250) — compile-verified, matrix operand order checked by reading, not run on hardware
//
#include <hip/hip_runtime.h>

// Problem constants (match reference)
#define B_  128
#define S_  64      // = d (channel dim)
#define N_  512
#define D_  64
#define THRESH 0.1f

// padded LDS row strides (in f16 elements); +8 f16 = +16B keeps b128 alignment
// and spreads the 16-lane fragment gathers across distinct LDS banks.
#define HP_  (D_ + 8)    // h / Wt row stride   (72 -> 144B, bank stride 36)
#define HWP_ (N_ + 8)    // hw_t row stride     (520 -> 1040B, bank stride 260%64=4)

typedef __attribute__((ext_vector_type(16))) _Float16 v16h;
typedef __attribute__((ext_vector_type(8)))  float    v8f;
typedef __attribute__((ext_vector_type(2)))  _Float16 h2;

__device__ __forceinline__ float sigmoidf_(float x) {
#if __has_builtin(__builtin_amdgcn_tanhf)
    return 0.5f + 0.5f * __builtin_amdgcn_tanhf(0.5f * x);   // v_tanh_f32: 1 trans op
#elif __has_builtin(__builtin_amdgcn_tanh_f32)
    return 0.5f + 0.5f * __builtin_amdgcn_tanh_f32(0.5f * x);
#else
    return 1.0f / (1.0f + __expf(-x));
#endif
}

// ---------------------------------------------------------------------------
// Fused attention vectors:
//   vq[c] = sum_c' Wq[c,c'] * wa[c'],  vk[c] = sum_c' Wk[c,c'] * wa[d+c']
//   scal[0] = bq . wa[:d] ; scal[1] = bk . wa[d:] + ba
// ---------------------------------------------------------------------------
__global__ void prep_kernel(const float* __restrict__ Wq, const float* __restrict__ bq,
                            const float* __restrict__ Wk, const float* __restrict__ bk,
                            const float* __restrict__ wa, const float* __restrict__ ba,
                            float* __restrict__ vq, float* __restrict__ vk,
                            float* __restrict__ scal) {
    int t = threadIdx.x;
    if (t < D_) {
        float aq = 0.f, ak = 0.f;
        for (int c = 0; c < D_; ++c) {
            aq += Wq[t * D_ + c] * wa[c];
            ak += Wk[t * D_ + c] * wa[D_ + c];
        }
        vq[t] = aq;
        vk[t] = ak;
    }
    if (t == 0) {
        float cq = 0.f, ck = 0.f;
        for (int c = 0; c < D_; ++c) {
            cq += bq[c] * wa[c];
            ck += bk[c] * wa[D_ + c];
        }
        scal[0] = cq;
        scal[1] = ck + ba[0];
    }
}

// ---------------------------------------------------------------------------
// h0[b,n,c] = (f16) x[b,c,n]
// ---------------------------------------------------------------------------
__global__ void transpose_in_kernel(const float* __restrict__ x, _Float16* __restrict__ h0) {
    int idx = blockIdx.x * blockDim.x + threadIdx.x;
    if (idx >= B_ * N_ * D_) return;
    int c = idx % D_;
    int n = (idx / D_) % N_;
    int b = idx / (D_ * N_);
    h0[idx] = (_Float16)x[(size_t)b * S_ * N_ + (size_t)c * N_ + n];
}

// ---------------------------------------------------------------------------
// Fully fused AdaGCN layer: one workgroup (8 wave32) per batch.
//   stage h[b] (f16) -> LDS via global_load_async_to_lds_b128 (ASYNCcnt)
//   sq/sk matvecs (q/k GEMMs algebraically collapsed), deg/dinv reduction,
//   hw = h @ W via v_wmma_f32_16x16x32_f16 (stored transposed+padded in LDS),
//   out = A_hat^T @ hw with the adjacency fragment synthesized in registers.
// ---------------------------------------------------------------------------
__global__ void __launch_bounds__(256)
layer_kernel(const _Float16* __restrict__ h,     // [B,N,D] f16
             const float* __restrict__ weight,   // [D,D]
             const float* __restrict__ bias,     // [D]
             const float* __restrict__ vq,       // [D]
             const float* __restrict__ vk,       // [D]
             const float* __restrict__ scal,     // [2]
             _Float16* __restrict__ hout16,      // [B,N,D] f16 (layers 0,1)
             float* __restrict__ out32,          // [B,S,N] f32 (last layer)
             int last)
{
    extern __shared__ char smem[];
    _Float16* h_lds  = (_Float16*)smem;                    // N * HP_ (async dest, 16B rows)
    _Float16* hw_t   = h_lds + (size_t)N_ * HP_;           // D * HWP_
    _Float16* Wt_lds = hw_t + (size_t)D_ * HWP_;           // D * HP_ : Wt[n*HP_+c] = W[c,n]
    float* sq_lds    = (float*)(Wt_lds + (size_t)D_ * HP_);
    float* sk_lds    = sq_lds + N_;
    float* dv_lds    = sk_lds + N_;
    float* bias_lds  = dv_lds + N_;
    float* vq_lds    = bias_lds + D_;
    float* vk_lds    = vq_lds + D_;

    const int b    = blockIdx.x;
    const int tid  = threadIdx.x;
    const int wave = tid >> 5;
    const int lane = tid & 31;

    // ---- stage h[b] into LDS with async b128 copies (row = 128B = 8 chunks) ----
    {
        const char* gsrc = (const char*)(h + (size_t)b * N_ * D_);
        unsigned ldsbase = (unsigned)(size_t)(void*)h_lds;
        for (int idx = tid; idx < N_ * 8; idx += 256) {
            int n  = idx >> 3;
            int ch = (idx & 7) * 16;
            unsigned    l = ldsbase + (unsigned)(n * (HP_ * 2) + ch);
            const char* g = gsrc + n * (D_ * 2) + ch;
            asm volatile("global_load_async_to_lds_b128 %0, %1, off"
                         :: "v"(l), "v"(g) : "memory");
        }
    }

    // ---- overlap: stage small operands with normal loads ----
    for (int idx = tid; idx < D_ * D_; idx += 256) {
        int c = idx / D_, n = idx % D_;
        Wt_lds[n * HP_ + c] = (_Float16)weight[idx];
    }
    for (int idx = tid; idx < D_; idx += 256) {
        bias_lds[idx] = bias[idx];
        vq_lds[idx]   = vq[idx];
        vk_lds[idx]   = vk[idx];
    }
    const float s0 = scal[0];
    const float s1 = scal[1];

    asm volatile("s_wait_asynccnt 0x0" ::: "memory");
    __syncthreads();

    // ---- sq/sk matvecs from LDS f16 ----
    for (int n = tid; n < N_; n += 256) {
        float aq = 0.f, ak = 0.f;
#pragma unroll 8
        for (int c = 0; c < D_; ++c) {
            float v = (float)h_lds[n * HP_ + c];
            aq += v * vq_lds[c];
            ak += v * vk_lds[c];
        }
        sq_lds[n] = aq + s0;
        sk_lds[n] = ak + s1;
    }
    __syncthreads();

    // ---- degree / inverse-sqrt normalization ----
    for (int j = tid; j < N_; j += 256) {
        float skj = sk_lds[j];
        float deg = 0.f;
        for (int i = 0; i < N_; ++i) {
            float p = sigmoidf_(sq_lds[i] + skj);
            if (p > THRESH && i != j) deg += p;
        }
        dv_lds[j] = (deg > 0.f) ? rsqrtf(deg) : 0.f;
    }
    __syncthreads();

    // per-lane fragment geometry (16-bit 16x32 A layout, wave32)
    const int m   = lane & 15;
    const int kb0 = (lane < 16) ? 0 : 8;
    const int kb1 = (lane < 16) ? 16 : 24;

    // ---- phase 1: hw = h @ W  -> hw_t[c*HWP_ + i] (f16, transposed) ----
    for (int t = wave; t < (N_ / 16) * (D_ / 16); t += 8) {
        int itile = t >> 2;
        int ctile = t & 3;
        v8f acc = {};
#pragma unroll
        for (int kk = 0; kk < D_; kk += 32) {
            v16h af, bf;
#pragma unroll
            for (int v = 0; v < 8; ++v) {
                int k0 = kk + ((v < 4) ? (kb0 + 2 * v) : (kb1 + 2 * (v - 4)));
                h2 pa = *(const h2*)&h_lds[(itile * 16 + m) * HP_ + k0];
                af[2 * v] = pa.x; af[2 * v + 1] = pa.y;
                h2 pb = *(const h2*)&Wt_lds[(ctile * 16 + m) * HP_ + k0];
                bf[2 * v] = pb.x; bf[2 * v + 1] = pb.y;
            }
            acc = __builtin_amdgcn_wmma_f32_16x16x32_f16(false, af, false, bf,
                                                         (short)0, acc, false, false);
        }
#pragma unroll
        for (int r = 0; r < 8; ++r) {
            int i_loc = itile * 16 + ((lane < 16) ? r : r + 8);
            int c_g   = ctile * 16 + m;
            hw_t[c_g * HWP_ + i_loc] = (_Float16)acc[r];
        }
    }
    __syncthreads();

    // ---- phase 2: out = A_hat^T @ hw (adjacency fragment built in registers) ----
    for (int jt = wave; jt < N_ / 16; jt += 8) {
        const int   j_row = jt * 16 + m;
        const float skj   = sk_lds[j_row];
        v8f acc[4] = {{}, {}, {}, {}};

        for (int ib = 0; ib < N_; ib += 32) {
            v16h af;
#pragma unroll
            for (int v = 0; v < 8; ++v) {
#pragma unroll
                for (int p = 0; p < 2; ++p) {
                    int k = (v < 4) ? (kb0 + 2 * v + p) : (kb1 + 2 * (v - 4) + p);
                    int i = ib + k;
                    float pr = sigmoidf_(sq_lds[i] + skj);
                    float av = (pr > THRESH && i != j_row) ? pr * dv_lds[i] : 0.0f;
                    af[2 * v + p] = (_Float16)av;
                }
            }
#pragma unroll
            for (int nt = 0; nt < 4; ++nt) {
                v16h bf;
#pragma unroll
                for (int v = 0; v < 8; ++v) {
                    int k0 = (v < 4) ? (kb0 + 2 * v) : (kb1 + 2 * (v - 4));
                    h2 pb = *(const h2*)&hw_t[(nt * 16 + m) * HWP_ + ib + k0];
                    bf[2 * v] = pb.x; bf[2 * v + 1] = pb.y;
                }
                acc[nt] = __builtin_amdgcn_wmma_f32_16x16x32_f16(false, af, false, bf,
                                                                 (short)0, acc[nt],
                                                                 false, false);
            }
        }

        // epilogue: scale by dinv[j], add bias; last layer writes f32 transposed
#pragma unroll
        for (int r = 0; r < 8; ++r) {
            int   j  = jt * 16 + ((lane < 16) ? r : r + 8);
            float dj = dv_lds[j];
#pragma unroll
            for (int nt = 0; nt < 4; ++nt) {
                int   c   = nt * 16 + m;
                float val = dj * acc[nt][r] + bias_lds[c];
                if (!last)
                    hout16[(size_t)b * N_ * D_ + (size_t)j * D_ + c] = (_Float16)val;
                else
                    out32[(size_t)b * S_ * N_ + (size_t)c * N_ + j] = val;
            }
        }
    }
}

// ---------------------------------------------------------------------------
extern "C" void kernel_launch(void* const* d_in, const int* in_sizes, int n_in,
                              void* d_out, int out_size, void* d_ws, size_t ws_size,
                              hipStream_t stream) {
    const float* x      = (const float*)d_in[0];
    const float* weight = (const float*)d_in[1];
    const float* bias   = (const float*)d_in[2];
    const float* Wq     = (const float*)d_in[3];
    const float* bq     = (const float*)d_in[4];
    const float* Wk     = (const float*)d_in[5];
    const float* bk     = (const float*)d_in[6];
    const float* wa     = (const float*)d_in[7];
    const float* ba     = (const float*)d_in[8];

    _Float16* hA  = (_Float16*)d_ws;                       // B*N*D f16
    _Float16* hB  = hA + (size_t)B_ * N_ * D_;             // B*N*D f16
    float*    vq  = (float*)(hB + (size_t)B_ * N_ * D_);   // D
    float*    vk  = vq + D_;                               // D
    float*    scal = vk + D_;                              // 2

    prep_kernel<<<1, 64, 0, stream>>>(Wq, bq, Wk, bk, wa, ba, vq, vk, scal);

    transpose_in_kernel<<<(B_ * N_ * D_ + 255) / 256, 256, 0, stream>>>(x, hA);

    size_t lds_bytes = (size_t)(N_ * HP_ + D_ * HWP_ + D_ * HP_) * sizeof(_Float16)
                     + (size_t)(3 * N_ + 3 * D_) * sizeof(float);

    _Float16* cur = hA;
    _Float16* nxt = hB;
    for (int L = 0; L < 3; ++L) {
        int last = (L == 2);
        layer_kernel<<<B_, 256, lds_bytes, stream>>>(
            cur, weight, bias, vq, vk, scal,
            nxt, (float*)d_out, last);
        _Float16* tmp = cur; cur = nxt; nxt = tmp;
    }
}
